// FunctionalAttention_18133351924468
// MI455X (gfx1250) — matrix operand order repaired and checked by reading, not compiled, not run on hardware
//
#include <hip/hip_runtime.h>

typedef _Float16 v16h __attribute__((ext_vector_type(16)));
typedef _Float16 v8h  __attribute__((ext_vector_type(8)));
typedef float    v8f  __attribute__((ext_vector_type(8)));

#define SEQ 2048
#define HD  64
#define NH  16
#define NB  2
#define BQ  128               // query rows per block (8 waves x 16)
#define BK  64                // key columns per tile
#define KP  (HD + 8)          // K LDS pitch in halves (144B rows, 16B aligned)
#define VP  (BK + 8)          // V LDS pitch in halves
#define PP  (BK + 8)          // P LDS pitch in halves
#define QSCALE 0.0225421100139f   // (1/64) * log2(e): base-2 softmax, scale folded into Q

// ---- DPP16 butterfly reduction within each 16-lane half (pure VALU, no LDS) ----
// xor1 = quad_perm[1,0,3,2], xor2 = quad_perm[2,3,0,1], xor7 = row_half_mirror,
// xor15 = row_mirror. {1,2,7,15} generates the full group of 16, so all lanes
// end up with the complete row reduction, confined to their 16-lane half.
template <int CTRL>
__device__ __forceinline__ float dppMov(float x) {
  return __int_as_float(__builtin_amdgcn_mov_dpp(__float_as_int(x), CTRL, 0xF, 0xF, true));
}
__device__ __forceinline__ float rowMax16(float x) {
  x = fmaxf(x, dppMov<0xB1>(x));
  x = fmaxf(x, dppMov<0x4E>(x));
  x = fmaxf(x, dppMov<0x141>(x));
  x = fmaxf(x, dppMov<0x140>(x));
  return x;
}
__device__ __forceinline__ float rowSum16(float x) {
  x += dppMov<0xB1>(x);
  x += dppMov<0x4E>(x);
  x += dppMov<0x141>(x);
  x += dppMov<0x140>(x);
  return x;
}

__global__ __launch_bounds__(256)
void fa_fwd_kernel(const float* __restrict__ q, const float* __restrict__ k,
                   const float* __restrict__ v, float* __restrict__ out) {
  __shared__ _Float16 Klds[BK * KP];       // [key][d]
  __shared__ _Float16 Vlds[HD * VP];       // [d][key] (transposed)
  __shared__ _Float16 Plds[8 * 16 * PP];   // per-wave P staging [wave][row][key]

  const int tid     = threadIdx.x;
  const int lane    = tid & 31;
  const int wave    = tid >> 5;
  const int l16     = lane & 15;
  const int halfSel = lane >> 4;
  const int h       = blockIdx.y;
  const int b       = blockIdx.z;
  const int qBase   = blockIdx.x * BQ;
  const int qRow0   = qBase + wave * 16;

  const size_t bhOff = (size_t)b * SEQ * HD * NH + h;
  const float* qb = q   + bhOff;   // elem (c,d) at [(c*HD+d)*NH]
  const float* kb = k   + bhOff;
  const float* vb = v   + bhOff;
  float*       ob = out + bhOff;

  // Cooperative-load coordinates: each thread owns 16 contiguous LDS halves,
  // so the LDS store side is 2x ds_store_b128 of packed f16.
  const int kKey  = tid >> 2;          // 0..63
  const int kD0   = (tid & 3) * 16;    // 0,16,32,48
  const int vD    = tid >> 2;
  const int vKey0 = (tid & 3) * 16;

  // ---- Q A-fragments (f16), scale*log2e folded in ----
  // A 16x32: lanes 0-15 row=l16, halves 0..7 -> K=0..7, 8..15 -> K=16..23
  //          lanes 16-31 row=l16, halves 0..7 -> K=8..15, 8..15 -> K=24..31
  v16h qA[2];
  {
    const float* qr = qb + (size_t)(qRow0 + l16) * HD * NH;
#pragma unroll
    for (int c = 0; c < 2; ++c) {
      const int dBase = c * 32 + halfSel * 8;
#pragma unroll
      for (int i = 0; i < 8; ++i) {
        qA[c][i]     = (_Float16)(qr[(dBase + i) * NH] * QSCALE);
        qA[c][i + 8] = (_Float16)(qr[(dBase + 16 + i) * NH] * QSCALE);
      }
    }
  }

  v8f acc[4];
#pragma unroll
  for (int cc = 0; cc < 4; ++cc)
#pragma unroll
    for (int r = 0; r < 8; ++r) acc[cc][r] = 0.0f;
  float m[8], l[8];
#pragma unroll
  for (int r = 0; r < 8; ++r) { m[r] = -__builtin_inff(); l[r] = 0.0f; }

  const int sRow0  = __builtin_amdgcn_readfirstlane(qRow0);  // scalar branch conds
  const int myLast = sRow0 + 15;
  const int nTiles = (qBase + BQ) / BK;

  // ---- software pipeline: tile t's K/V staged in regs during tile t-1 compute ----
  float kr[16], vr[16];
#pragma unroll
  for (int i = 0; i < 16; ++i)
    kr[i] = kb[((size_t)kKey * HD + kD0 + i) * NH];
#pragma unroll
  for (int i = 0; i < 16; ++i)
    vr[i] = vb[((size_t)(vKey0 + i) * HD + vD) * NH];

  for (int t = 0; t < nTiles; ++t) {
    const int jBase = t * BK;
    __syncthreads();   // all waves done reading previous LDS tile
    {  // staged regs -> LDS as packed f16 (2x b128 each for K and V)
      v8h a0, a1;
#pragma unroll
      for (int i = 0; i < 8; ++i) { a0[i] = (_Float16)kr[i]; a1[i] = (_Float16)kr[i + 8]; }
      _Float16* kd = &Klds[kKey * KP + kD0];
      *(v8h*)kd = a0; *(v8h*)(kd + 8) = a1;
#pragma unroll
      for (int i = 0; i < 8; ++i) { a0[i] = (_Float16)vr[i]; a1[i] = (_Float16)vr[i + 8]; }
      _Float16* vd = &Vlds[vD * VP + vKey0];
      *(v8h*)vd = a0; *(v8h*)(vd + 8) = a1;
    }
    if (t + 1 < nTiles) {  // issue next tile's gathers; latency hidden behind compute
      const int jN = jBase + BK;
#pragma unroll
      for (int i = 0; i < 16; ++i)
        kr[i] = kb[((size_t)(jN + kKey) * HD + kD0 + i) * NH];
#pragma unroll
      for (int i = 0; i < 16; ++i)
        vr[i] = vb[((size_t)(jN + vKey0 + i) * HD + vD) * NH];
    }
    __syncthreads();   // LDS tile ready

    if (jBase > myLast) continue;   // scalar skip: EXEC stays all-ones for WMMA

    // ---- S = Q K^T (base-2 logits); c-outer so the 4 accum chains interleave ----
    v8f S[4];
#pragma unroll
    for (int t2 = 0; t2 < 4; ++t2)
#pragma unroll
      for (int r = 0; r < 8; ++r) S[t2][r] = 0.0f;
#pragma unroll
    for (int c = 0; c < 2; ++c) {
#pragma unroll
      for (int t2 = 0; t2 < 4; ++t2) {
        // B 32x16: lane<16 col=l16 rows d=c*32+0..7,16..23; lane>=16 rows 8..15,24..31
        const _Float16* base = &Klds[(t2 * 16 + l16) * KP + c * 32 + halfSel * 8];
        v8h lo = *(const v8h*)base;
        v8h hi = *(const v8h*)(base + 16);
        v16h bf;
#pragma unroll
        for (int i = 0; i < 8; ++i) { bf[i] = lo[i]; bf[i + 8] = hi[i]; }
        S[t2] = __builtin_amdgcn_wmma_f32_16x16x32_f16(false, qA[c], false, bf,
                                                       (short)0, S[t2], false, false);
      }
    }

    // ---- causal mask only on the diagonal tile (scalar condition) ----
    if (jBase + BK - 1 > sRow0) {
#pragma unroll
      for (int t2 = 0; t2 < 4; ++t2) {
        const int col = jBase + t2 * 16 + l16;
#pragma unroll
        for (int r = 0; r < 8; ++r) {
          const int row = qRow0 + r + 8 * halfSel;
          if (col > row) S[t2][r] = -__builtin_inff();
        }
      }
    }

    // ---- online softmax (base-2), DPP reductions ----
    float mt[8], alpha[8], lt[8];
#pragma unroll
    for (int r = 0; r < 8; ++r)
      mt[r] = fmaxf(fmaxf(S[0][r], S[1][r]), fmaxf(S[2][r], S[3][r]));
#pragma unroll
    for (int r = 0; r < 8; ++r) mt[r] = rowMax16(mt[r]);
#pragma unroll
    for (int r = 0; r < 8; ++r) {
      const float mn = fmaxf(m[r], mt[r]);
      alpha[r] = __builtin_amdgcn_exp2f(m[r] - mn);
      m[r] = mn;
#pragma unroll
      for (int t2 = 0; t2 < 4; ++t2) S[t2][r] = __builtin_amdgcn_exp2f(S[t2][r] - mn);
      lt[r] = (S[0][r] + S[1][r]) + (S[2][r] + S[3][r]);
    }
#pragma unroll
    for (int r = 0; r < 8; ++r) lt[r] = rowSum16(lt[r]);
#pragma unroll
    for (int r = 0; r < 8; ++r) l[r] = l[r] * alpha[r] + lt[r];
#pragma unroll
    for (int cc = 0; cc < 4; ++cc)
#pragma unroll
      for (int r = 0; r < 8; ++r) acc[cc][r] *= alpha[r];

    // ---- P: f32 C-layout -> f16 A-layout via per-wave LDS staging ----
    _Float16* pw = &Plds[wave * 16 * PP];
#pragma unroll
    for (int t2 = 0; t2 < 4; ++t2)
#pragma unroll
      for (int r = 0; r < 8; ++r)
        pw[(r + 8 * halfSel) * PP + t2 * 16 + l16] = (_Float16)S[t2][r];
    v16h pA[2];
#pragma unroll
    for (int kc = 0; kc < 2; ++kc) {
      const _Float16* base = &pw[l16 * PP + kc * 32 + halfSel * 8];
      v8h lo = *(const v8h*)base;          // keys kc*32 + halfSel*8 + 0..7
      v8h hi = *(const v8h*)(base + 16);   // keys kc*32 + 16 + halfSel*8 + 0..7
#pragma unroll
      for (int i = 0; i < 8; ++i) { pA[kc][i] = lo[i]; pA[kc][i + 8] = hi[i]; }
    }

    // ---- acc += P * V; kc-outer so the 4 acc chains interleave ----
#pragma unroll
    for (int kc = 0; kc < 2; ++kc) {
#pragma unroll
      for (int cc = 0; cc < 4; ++cc) {
        // B 32x16: lane<16 col d=cc*16+l16 rows keys kc*32+0..7,16..23; lane>=16 keys +8..15,24..31
        const _Float16* base = &Vlds[(cc * 16 + l16) * VP + kc * 32 + halfSel * 8];
        v8h lo = *(const v8h*)base;
        v8h hi = *(const v8h*)(base + 16);
        v16h vf;
#pragma unroll
        for (int i = 0; i < 8; ++i) { vf[i] = lo[i]; vf[i + 8] = hi[i]; }
        acc[cc] = __builtin_amdgcn_wmma_f32_16x16x32_f16(false, pA[kc], false, vf,
                                                         (short)0, acc[cc], false, false);
      }
    }
  }

  // ---- normalize + store (C/D layout: VGPR r, lanes0-15 row r, lanes16-31 row r+8) ----
  float inv[8];
#pragma unroll
  for (int r = 0; r < 8; ++r) inv[r] = 1.0f / l[r];
#pragma unroll
  for (int cc = 0; cc < 4; ++cc) {
#pragma unroll
    for (int r = 0; r < 8; ++r) {
      const int row = qRow0 + r + 8 * halfSel;
      const int d   = cc * 16 + l16;
      ob[((size_t)row * HD + d) * NH] = acc[cc][r] * inv[r];
    }
  }
}

extern "C" void kernel_launch(void* const* d_in, const int* in_sizes, int n_in,
                              void* d_out, int out_size, void* d_ws, size_t ws_size,
                              hipStream_t stream) {
  (void)in_sizes; (void)n_in; (void)out_size; (void)d_ws; (void)ws_size;
  const float* q = (const float*)d_in[0];
  const float* k = (const float*)d_in[1];
  const float* v = (const float*)d_in[2];
  float* out = (float*)d_out;
  dim3 grid(SEQ / BQ, NH, NB);   // (16, 16, 2)
  dim3 block(256);               // 8 waves
  fa_fwd_kernel<<<grid, block, 0, stream>>>(q, k, v, out);
}
